// Contrast_75866302316823
// MI455X (gfx1250) — compile-verified, matrix operand order checked by reading, not run on hardware
//
#include <hip/hip_runtime.h>
#include <hip/hip_bf16.h>
#include <math.h>

#define NROWS 8192
#define HDIM  512
#define TAU_INV 2.0f
#define EPSV 1e-8f

typedef float   v2f   __attribute__((ext_vector_type(2)));
typedef float   v8f   __attribute__((ext_vector_type(8)));
typedef __bf16  v8bf  __attribute__((ext_vector_type(8)));
typedef __bf16  v16bf __attribute__((ext_vector_type(16)));
typedef unsigned int u32x4 __attribute__((ext_vector_type(4)));
typedef int          i32x4 __attribute__((ext_vector_type(4)));
typedef int          i32x8 __attribute__((ext_vector_type(8)));

#define BM 128
#define BN 128
#define BK 32

// ---------------------------------------------------------------------------
// TDM: DMA a 2D bf16 tile [BN rows x BK k-elems] of the [NROWS x HDIM] tensor
// into LDS at byte offset lds_off. D# built per CDNA5 ISA ch.8 (group0/group1
// bit layout); groups 2/3 zero (2D tensor). Tracked by TENSORcnt.
// 6-arg builtin form (clang-23 / therock-10.0 headers).
// ---------------------------------------------------------------------------
__device__ __forceinline__ void tdm_load_tile(const void* gsrc, unsigned lds_off)
{
    const unsigned long long ga = (unsigned long long)(size_t)gsrc;
    u32x4 g0;
    g0[0] = 1u;                                      // count=1, user descriptor
    g0[1] = lds_off;                                 // lds_addr (bytes)
    g0[2] = (unsigned)ga;                            // global_addr[31:0]
    g0[3] = (unsigned)((ga >> 32) & 0x01FFFFFFu)     // global_addr[56:32]
          | (2u << 30);                              // type = 2 ("image")
    i32x8 g1;
    g1[0] = 1 << 16;                                 // data_size = 1 -> 2 bytes
    g1[1] = (int)(((unsigned)HDIM)  << 16);          // tensor_dim0[15:0] @ bits63:48
    g1[2] = (int)(((unsigned)NROWS) << 16);          // dim0 hi=0 | tensor_dim1[15:0]
    g1[3] = (int)(((unsigned)BK)    << 16);          // dim1 hi=0 | tile_dim0
    g1[4] = BN;                                      // tile_dim1 | tile_dim2=0
    g1[5] = HDIM;                                    // tensor_dim0_stride[31:0]
    g1[6] = 0;                                       // stride hi | dim1_stride lo
    g1[7] = 0;
    i32x4 z4 = (i32x4){0, 0, 0, 0};
    i32x8 z8 = (i32x8){0, 0, 0, 0, 0, 0, 0, 0};
    __builtin_amdgcn_tensor_load_to_lds(g0, g1, z4, z4, z8, 0);
}

// ---------------------------------------------------------------------------
// MLP GEMM: Y = X @ W^T + b, mode 0: ELU (hidden), mode 1: fp32 + bf16 out.
// fp32 WMMA 16x16x4. Block = 256 threads (8 waves), block tile 64x64,
// each wave computes a 16x32 strip (2 C tiles).
// ---------------------------------------------------------------------------
__global__ __launch_bounds__(256) void mlp_gemm_kernel(
    const float* __restrict__ X, const float* __restrict__ W,
    const float* __restrict__ bias,
    float* __restrict__ Yf, __bf16* __restrict__ Yb, int mode)
{
    const int tid  = threadIdx.x;
    const int lane = tid & 31;
    const int wave = tid >> 5;
    const int l15  = lane & 15;
    const int hi   = lane >> 4;           // 0 or 1

    const int mbase = blockIdx.x * 64 + (wave & 3) * 16;
    const int nbase = blockIdx.y * 64 + (wave >> 2) * 32;

    v8f acc0 = {}; v8f acc1 = {};

    const int am  = mbase + l15;
    const int bn0 = nbase + l15;
    const int bn1 = nbase + 16 + l15;

    for (int kk = 0; kk < HDIM; kk += 4) {
        const int kp = kk + hi * 2;   // ISA f32 A 16x4 / B 4x16 lane layout
        v2f a  = *reinterpret_cast<const v2f*>(X + (size_t)am  * HDIM + kp);
        v2f b0 = *reinterpret_cast<const v2f*>(W + (size_t)bn0 * HDIM + kp);
        v2f b1 = *reinterpret_cast<const v2f*>(W + (size_t)bn1 * HDIM + kp);
        acc0 = __builtin_amdgcn_wmma_f32_16x16x4_f32(false, a, false, b0,
                                                     (short)0, acc0, false, false);
        acc1 = __builtin_amdgcn_wmma_f32_16x16x4_f32(false, a, false, b1,
                                                     (short)0, acc1, false, false);
    }

    // C layout: VGPR v -> row mbase + v + hi*8 ; col = tilebase + l15
    #pragma unroll
    for (int t = 0; t < 2; ++t) {
        v8f acc = t ? acc1 : acc0;
        const int n  = nbase + t * 16 + l15;
        const float bn = bias[n];
        #pragma unroll
        for (int v = 0; v < 8; ++v) {
            const int row = mbase + hi * 8 + v;
            float val = acc[v] + bn;
            if (mode == 0) {
                val = val > 0.f ? val : expm1f(val);   // ELU
                Yf[(size_t)row * HDIM + n] = val;
            } else {
                Yf[(size_t)row * HDIM + n] = val;
                Yb[(size_t)row * HDIM + n] = (__bf16)val;
            }
        }
    }
}

// ---------------------------------------------------------------------------
// Reciprocal row L2 norms of Z [NROWS, HDIM]
// ---------------------------------------------------------------------------
__global__ __launch_bounds__(256) void rownorm_kernel(
    const float* __restrict__ Z, float* __restrict__ rn)
{
    __shared__ float red[8];
    const int row = blockIdx.x;
    const int tid = threadIdx.x;
    float s = 0.f;
    for (int j = tid; j < HDIM; j += 256) {
        float v = Z[(size_t)row * HDIM + j];
        s += v * v;
    }
    for (int off = 16; off; off >>= 1) s += __shfl_xor(s, off, 32);
    if ((tid & 31) == 0) red[tid >> 5] = s;
    __syncthreads();
    if (tid < 8) {
        float t = red[tid];
        for (int off = 4; off; off >>= 1) t += __shfl_xor(t, off, 8);
        if (tid == 0) rn[row] = 1.0f / sqrtf(t);
    }
}

// ---------------------------------------------------------------------------
// Similarity GEMM (bf16 WMMA 16x16x32) + fused exp / row reductions.
// Block tile 128x128, 8 waves; wave w owns rows [bx*128+w*16, +16) across the
// 128-col window. B panel (z1 rows) DMA'd into double-buffered LDS by the TDM
// (wave 0 issues; s_wait_tensorcnt 1 overlaps next DMA with current compute).
// Writes per-(column-block) partial row sums -> deterministic (no atomics).
// ---------------------------------------------------------------------------
__global__ __launch_bounds__(256) void sim_kernel(
    const __bf16* __restrict__ Z0b, const __bf16* __restrict__ Z1b,
    const float* __restrict__ rn0, const float* __restrict__ rn1,
    const float* __restrict__ pos,
    float* __restrict__ rs_part, float* __restrict__ p_part)
{
    __shared__ __bf16 Bs[2][BN * BK];   // 2 x (128 rows(n) x 32 k) = 16 KB

    const int tid  = threadIdx.x;
    const int lane = tid & 31;
    const int wave = tid >> 5;
    const int l15  = lane & 15;
    const int hi   = lane >> 4;

    const int mbase  = blockIdx.x * BM + wave * 16;
    const int nblock = blockIdx.y * BN;

    v8f acc[8];
    #pragma unroll
    for (int t = 0; t < 8; ++t) acc[t] = (v8f){};

    const size_t arow  = (size_t)(mbase + l15) * HDIM;
    const __bf16* Bsrc = Z1b + (size_t)nblock * HDIM;
    // flat->LDS offset: low 32 bits of the shared-aperture address
    const unsigned ldsoff0 = (unsigned)(size_t)(&Bs[0][0]);
    const unsigned ldsoff1 = (unsigned)(size_t)(&Bs[1][0]);

    const int NCHUNK = HDIM / BK;       // 16
    if (wave == 0) tdm_load_tile(Bsrc, ldsoff0);

    for (int ic = 0; ic < NCHUNK; ++ic) {
        const int kk = ic * BK;
        if (wave == 0) {
            if (ic + 1 < NCHUNK) {
                // prefetch next chunk into the other buffer (read last iter,
                // WAR protected by the trailing barrier of the last iter)
                tdm_load_tile(Bsrc + (size_t)(kk + BK),
                              ((ic + 1) & 1) ? ldsoff1 : ldsoff0);
                __builtin_amdgcn_s_wait_tensorcnt(1);   // current chunk done
            } else {
                __builtin_amdgcn_s_wait_tensorcnt(0);
            }
        }
        __syncthreads();                 // LDS panel visible to all waves

        const __bf16* Bcur = &Bs[ic & 1][0];

        // A fragment: ISA 16-bit A 16x32 layout -> chunks at hi*8 and 16+hi*8
        v8bf a0 = *reinterpret_cast<const v8bf*>(Z0b + arow + kk + hi * 8);
        v8bf a1 = *reinterpret_cast<const v8bf*>(Z0b + arow + kk + 16 + hi * 8);
        v16bf a;
        #pragma unroll
        for (int i = 0; i < 8; ++i) { a[i] = a0[i]; a[i + 8] = a1[i]; }

        #pragma unroll
        for (int t = 0; t < 8; ++t) {
            const int n = t * 16 + l15;
            v8bf b0 = *reinterpret_cast<const v8bf*>(&Bcur[n * BK + hi * 8]);
            v8bf b1 = *reinterpret_cast<const v8bf*>(&Bcur[n * BK + 16 + hi * 8]);
            v16bf b;
            #pragma unroll
            for (int i = 0; i < 8; ++i) { b[i] = b0[i]; b[i + 8] = b1[i]; }
            acc[t] = __builtin_amdgcn_wmma_f32_16x16x32_bf16(
                false, a, false, b, (short)0, acc[t], false, false);
        }
        __syncthreads();                 // all reads done before buffer refill
    }

    // Fused epilogue: s = exp(dot * rn0[m] * rn1[n] / tau); reduce over n.
    float rs[8], ps[8];
    #pragma unroll
    for (int v = 0; v < 8; ++v) { rs[v] = 0.f; ps[v] = 0.f; }

    #pragma unroll
    for (int t = 0; t < 8; ++t) {
        const int n  = nblock + t * 16 + l15;
        const float r1 = rn1[n];
        #pragma unroll
        for (int v = 0; v < 8; ++v) {
            const int m = mbase + hi * 8 + v;
            const float s = __expf(acc[t][v] * rn0[m] * r1 * TAU_INV);
            rs[v] += s;
            ps[v] += s * pos[(size_t)m * NROWS + n];
        }
    }

    // cross-lane reduce within each 16-lane group (same m, n varies)
    #pragma unroll
    for (int v = 0; v < 8; ++v) {
        float a = rs[v], b = ps[v];
        for (int off = 8; off; off >>= 1) {
            a += __shfl_xor(a, off, 16);
            b += __shfl_xor(b, off, 16);
        }
        if (l15 == 0) {
            const int m = mbase + hi * 8 + v;
            rs_part[(size_t)blockIdx.y * NROWS + m] = a;
            p_part [(size_t)blockIdx.y * NROWS + m] = b;
        }
    }
}

// ---------------------------------------------------------------------------
// Final loss: reduce 64 column-block partials per row, then mean over rows.
// ---------------------------------------------------------------------------
__global__ __launch_bounds__(256) void loss_kernel(
    const float* __restrict__ rs_part, const float* __restrict__ p_part,
    float* __restrict__ out)
{
    __shared__ float red[8];
    const int tid = threadIdx.x;
    float acc = 0.f;
    for (int i = tid; i < NROWS; i += 256) {
        float rs = 0.f, pp = 0.f;
        for (int by = 0; by < NROWS / BN; ++by) {
            rs += rs_part[(size_t)by * NROWS + i];
            pp += p_part [(size_t)by * NROWS + i];
        }
        acc += logf(pp / (rs + EPSV) + EPSV);
    }
    for (int off = 16; off; off >>= 1) acc += __shfl_xor(acc, off, 32);
    if ((tid & 31) == 0) red[tid >> 5] = acc;
    __syncthreads();
    if (tid < 8) {
        float t = red[tid];
        for (int off = 4; off; off >>= 1) t += __shfl_xor(t, off, 8);
        if (tid == 0) *out = -t / (float)NROWS;
    }
}

// ---------------------------------------------------------------------------
extern "C" void kernel_launch(void* const* d_in, const int* in_sizes, int n_in,
                              void* d_out, int out_size, void* d_ws, size_t ws_size,
                              hipStream_t stream)
{
    const float* embd0 = (const float*)d_in[0];
    const float* embd1 = (const float*)d_in[1];
    const float* pos   = (const float*)d_in[2];
    const float* W1    = (const float*)d_in[3];
    const float* b1    = (const float*)d_in[4];
    const float* W2    = (const float*)d_in[5];
    const float* b2    = (const float*)d_in[6];

    float* out  = (float*)d_out;
    float* z0   = out;
    float* z1   = out + (size_t)NROWS * HDIM;
    float* loss = out + (size_t)2 * NROWS * HDIM;

    char* ws = (char*)d_ws;
    float*  h     = (float*)(ws);                                  // 16 MB
    __bf16* zb0   = (__bf16*)(ws + (size_t)16 * 1024 * 1024);      //  8 MB
    __bf16* zb1   = (__bf16*)(ws + (size_t)24 * 1024 * 1024);      //  8 MB
    float*  rn0   = (float*)(ws + (size_t)32 * 1024 * 1024);       // 32 KB
    float*  rn1   = rn0 + NROWS;                                   // 32 KB
    float*  rs_pt = (float*)(ws + (size_t)34 * 1024 * 1024);       //  2 MB
    float*  p_pt  = (float*)(ws + (size_t)36 * 1024 * 1024);       //  2 MB

    dim3 blk(256);
    dim3 gmm(NROWS / 64, HDIM / 64);

    // embd0 path: h = ELU(x W1^T + b1); z0 = h W2^T + b2 (+bf16 copy)
    mlp_gemm_kernel<<<gmm, blk, 0, stream>>>(embd0, W1, b1, h, (__bf16*)nullptr, 0);
    mlp_gemm_kernel<<<gmm, blk, 0, stream>>>(h, W2, b2, z0, zb0, 1);
    // embd1 path
    mlp_gemm_kernel<<<gmm, blk, 0, stream>>>(embd1, W1, b1, h, (__bf16*)nullptr, 0);
    mlp_gemm_kernel<<<gmm, blk, 0, stream>>>(h, W2, b2, z1, zb1, 1);

    rownorm_kernel<<<dim3(NROWS), blk, 0, stream>>>(z0, rn0);
    rownorm_kernel<<<dim3(NROWS), blk, 0, stream>>>(z1, rn1);

    sim_kernel<<<dim3(NROWS / BM, NROWS / BN), blk, 0, stream>>>(
        zb0, zb1, rn0, rn1, pos, rs_pt, p_pt);

    loss_kernel<<<dim3(1), blk, 0, stream>>>(rs_pt, p_pt, loss);
}
